// AttentionManifold_12206297055697
// MI455X (gfx1250) — matrix operand order repaired and checked by reading, not compiled, not run on hardware
//
#include <hip/hip_runtime.h>
#include <hip/hip_bf16.h>
#include <math.h>

// ============================================================================
// AttentionManifold on MI455X (gfx1250, wave32, WMMA)
//
// All GEMM-shaped stages run on v_wmma_f32_16x16x32_f16 (f16 in, f32 acc).
// GEMM operands are staged in LDS *pre-swizzled into WMMA fragment order*
// as packed f16, so each fragment load is one aligned 32B LDS vector read
// (no per-element ds_load + cvt + dscnt-wait serialization).
// 32x32 symmetric eigh = parallel cyclic Jacobi in f32 LDS (16 disjoint
// pairs/round, two-phase disjoint col/row rotations, 256 threads).
// ============================================================================

typedef __attribute__((ext_vector_type(16))) _Float16 v16h;
typedef __attribute__((ext_vector_type(8)))  float    v8f;

#define WMMA_F32_F16(a, b, c) \
  __builtin_amdgcn_wmma_f32_16x16x32_f16(false, (a), false, (b), (short)0, (c), false, false)

// ---- fragment index maps (this code's contiguous k order, same for A and B) --------------------
// A 16x32 f16: lane = (k>>4)*16 + m, elem e = k&15
__device__ __forceinline__ void a_map(int m, int kk, int& l, int& e) {
  l = ((kk >> 4) << 4) + m;
  e = kk & 15;
}
// B 32x16 f16: lane = (k>>4)*16 + n, elem e = k&15
__device__ __forceinline__ void b_map(int kk, int n, int& l, int& e) {
  l = ((kk >> 4) << 4) + n;
  e = kk & 15;
}
// B fragment directly from a global f32 row-major matrix (16-lane coalesced)
__device__ __forceinline__ v16h frag_b_g(const float* B, int ldb, int k0, int col0, int lane) {
  const int n = col0 + (lane & 15);
  const int h = (lane >> 4) & 1;
  v16h vb;
#pragma unroll
  for (int e = 0; e < 16; ++e)
    vb[e] = (_Float16)B[(size_t)(k0 + (h << 4) + e) * ldb + n];
  return vb;
}
// C/D 16x16 f32: lane l -> n = l%16, half h = l/16; reg r -> m = h*8 + r
__device__ __forceinline__ void store_c(float* C, int ldc, int row0, int col0, int lane, v8f c) {
  const int n = col0 + (lane & 15);
  const int h = (lane >> 4) & 1;
#pragma unroll
  for (int r = 0; r < 8; ++r)
    C[(row0 + (h << 3) + r) * ldc + n] = c[r];
}

// ---- parallel cyclic Jacobi eigh of a 32x32 symmetric matrix in LDS --------
// Requires blockDim.x == 256. Tournament pairing: 16 disjoint pairs/round,
// 31 rounds/sweep; rotations applied in two disjoint phases.
__device__ void jacobi32(float* S, float* U, float* ev,
                         int* pp, int* pq, float* rc, float* rs, int tid) {
  for (int idx = tid; idx < 32 * 32; idx += 256) {
    const int r = idx >> 5, c = idx & 31;
    U[r * 33 + c] = (r == c) ? 1.f : 0.f;
  }
  __syncthreads();
  for (int sweep = 0; sweep < 7; ++sweep) {
    for (int rnd = 0; rnd < 31; ++rnd) {
      if (tid < 16) {
        int p, q;
        if (tid == 0) { p = 31; q = rnd % 31; }
        else { p = (rnd + tid) % 31; q = (rnd + 31 - tid) % 31; }
        if (p > q) { const int t = p; p = q; q = t; }
        const float app = S[p * 33 + p];
        const float aqq = S[q * 33 + q];
        const float apq = S[p * 33 + q];
        float c, s;
        if (fabsf(apq) > 1e-12f * (fabsf(app) + fabsf(aqq)) + 1e-30f) {
          const float tau = (aqq - app) / (2.f * apq);
          const float t = copysignf(1.f, tau) / (fabsf(tau) + sqrtf(1.f + tau * tau));
          c = 1.f / sqrtf(1.f + t * t);
          s = t * c;
        } else { c = 1.f; s = 0.f; }
        pp[tid] = p; pq[tid] = q; rc[tid] = c; rs[tid] = s;
      }
      __syncthreads();
      for (int w = tid; w < 512; w += 256) {       // cols of S + U (A*J, U*J)
        const int pi = w >> 5, row = w & 31;
        const int p = pp[pi], q = pq[pi];
        const float c = rc[pi], s = rs[pi];
        const float sp = S[row * 33 + p], sq = S[row * 33 + q];
        S[row * 33 + p] = c * sp - s * sq;
        S[row * 33 + q] = s * sp + c * sq;
        const float up = U[row * 33 + p], uq = U[row * 33 + q];
        U[row * 33 + p] = c * up - s * uq;
        U[row * 33 + q] = s * up + c * uq;
      }
      __syncthreads();
      for (int w = tid; w < 512; w += 256) {       // rows of S (J^T * A)
        const int pi = w >> 5, col = w & 31;
        const int p = pp[pi], q = pq[pi];
        const float c = rc[pi], s = rs[pi];
        const float sp = S[p * 33 + col], sq = S[q * 33 + col];
        S[p * 33 + col] = c * sp - s * sq;
        S[q * 33 + col] = s * sp + c * sq;
      }
      __syncthreads();
    }
  }
  if (tid < 32) ev[tid] = S[tid * 33 + tid];
  __syncthreads();
}

// ============================================================================
// Kernel 1: per (b,m) matrix — bimap (WMMA) + matrix log (Jacobi + WMMA)
// grid = 4096, block = 256 (8 waves)
// ============================================================================
__global__ void __launch_bounds__(256) k1_bimap_log(
    const float* __restrict__ x,
    const float* __restrict__ Wq, const float* __restrict__ Wk, const float* __restrict__ Wv,
    float* __restrict__ logQ, float* __restrict__ logK, float* __restrict__ logV,
    float* __restrict__ q2, float* __restrict__ k2) {
  __shared__ v16h XA[4 * 2 * 32];   // X as A-frags [rt][kc][lane]
  __shared__ v16h WB[2 * 2 * 32];   // W as B-frags [kc][ct][lane]
  __shared__ v16h WAT[2 * 2 * 32];  // W^T as A-frags [rt][kc][lane]
  __shared__ v16h TB[2 * 2 * 32];   // T = X*W as B-frags [kc][ct][lane]
  __shared__ v16h GA[2 * 32];       // U*diag(log s) as A-frags [rt][lane]
  __shared__ v16h UB[2 * 32];       // U^T as B-frags [ct][lane]
  __shared__ float SS[32 * 33], US[32 * 33], LT[32 * 33];
  __shared__ float ev[32], fv[32];
  __shared__ float rc[16], rs[16];
  __shared__ int   pp[16], pq[16];
  __shared__ float red[256];

  _Float16* XAh  = (_Float16*)XA;
  _Float16* WBh  = (_Float16*)WB;
  _Float16* WATh = (_Float16*)WAT;
  _Float16* TBh  = (_Float16*)TB;
  _Float16* GAh  = (_Float16*)GA;
  _Float16* UBh  = (_Float16*)UB;

  const int tid = threadIdx.x;
  const int lane = tid & 31;
  const int wave = tid >> 5;
  const int mat = blockIdx.x;                // b*128 + m

  // stage X (64x64) into A-fragment layout, coalesced global reads
  for (int idx = tid; idx < 4096; idx += 256) {
    const int r = idx >> 6, c = idx & 63;
    int l, e; a_map(r & 15, c & 31, l, e);
    XAh[((((r >> 4) << 1) + (c >> 5)) * 32 + l) * 16 + e] =
        (_Float16)x[(size_t)mat * 4096 + idx];
  }

  for (int w = 0; w < 3; ++w) {
    const float* Wg = (w == 0) ? Wq : ((w == 1) ? Wk : Wv);
    float* dst = (w == 0) ? logQ : ((w == 1) ? logK : logV);

    // stage W (64x32) into both B-frag and (transposed) A-frag layouts
    for (int idx = tid; idx < 2048; idx += 256) {
      const int r = idx >> 5, c = idx & 31;    // r = k row (0..63), c = col (0..31)
      const float v = Wg[idx];
      int l, e;
      b_map(r & 31, c & 15, l, e);
      WBh[((((r >> 5) << 1) + (c >> 4)) * 32 + l) * 16 + e] = (_Float16)v;
      a_map(c & 15, r & 31, l, e);
      WATh[((((c >> 4) << 1) + (r >> 5)) * 32 + l) * 16 + e] = (_Float16)v;
    }
    __syncthreads();

    // T = X(64x64) * W(64x32): 8 tiles, one per wave; result scattered into
    // B-frag layout for the next GEMM (f16, matching previous precision)
    {
      const int rt = wave >> 1, ct = wave & 1;
      v8f acc = {};
#pragma unroll
      for (int kc = 0; kc < 2; ++kc)
        acc = WMMA_F32_F16(XA[((rt << 1) + kc) * 32 + lane],
                           WB[((kc << 1) + ct) * 32 + lane], acc);
      const int n = lane & 15, h = lane >> 4;
#pragma unroll
      for (int r = 0; r < 8; ++r) {
        const int row = rt * 16 + (h << 3) + r;        // 0..63 (k of next GEMM)
        int l, e; b_map(row & 31, n, l, e);
        TBh[((((row >> 5) << 1) + ct) * 32 + l) * 16 + e] = (_Float16)acc[r];
      }
    }
    __syncthreads();

    // S = W^T(32x64) * T(64x32): 4 tiles on waves 0..3 -> f32 LDS for Jacobi
    if (wave < 4) {
      const int rt = wave >> 1, ct = wave & 1;
      v8f acc = {};
#pragma unroll
      for (int kc = 0; kc < 2; ++kc)
        acc = WMMA_F32_F16(WAT[((rt << 1) + kc) * 32 + lane],
                           TB[((kc << 1) + ct) * 32 + lane], acc);
      store_c(SS, 33, rt * 16, ct * 16, lane, acc);
    }
    __syncthreads();

    jacobi32(SS, US, ev, pp, pq, rc, rs, tid);
    if (tid < 32) fv[tid] = logf(fmaxf(ev[tid], 1e-30f));
    __syncthreads();

    // build U*diag(log s) (A-frags) and U^T (B-frags) cooperatively
    for (int idx = tid; idx < 1024; idx += 256) {
      const int r = idx >> 5, c = idx & 31;
      const float u = US[r * 33 + c];
      int l, e;
      a_map(r & 15, c, l, e);                      // A[m=r][k=c] = U[r][c]*f[c]
      GAh[((r >> 4) * 32 + l) * 16 + e] = (_Float16)(u * fv[c]);
      b_map(c, r & 15, l, e);                      // B[k=c][n=r] = U[r][c]
      UBh[((r >> 4) * 32 + l) * 16 + e] = (_Float16)u;
    }
    __syncthreads();

    // log(S) = (U diag) * U^T: 4 tiles on waves 0..3, single K=32 chunk
    if (wave < 4) {
      const int rt = wave >> 1, ct = wave & 1;
      v8f acc = {};
      acc = WMMA_F32_F16(GA[rt * 32 + lane], UB[ct * 32 + lane], acc);
      store_c(LT, 33, rt * 16, ct * 16, lane, acc);
    }
    __syncthreads();

    // write log matrix + Frobenius^2 norm (for energy, Q and K only)
    float part = 0.f;
    for (int idx = tid; idx < 1024; idx += 256) {
      const float v = LT[(idx >> 5) * 33 + (idx & 31)];
      dst[(size_t)mat * 1024 + idx] = v;
      part += v * v;
    }
    red[tid] = part;
    __syncthreads();
    for (int sft = 128; sft > 0; sft >>= 1) {
      if (tid < sft) red[tid] += red[tid + sft];
      __syncthreads();
    }
    if (tid == 0) {
      if (w == 0) q2[mat] = red[0];
      else if (w == 1) k2[mat] = red[0];
    }
    __syncthreads();
  }
}

// ============================================================================
// Kernel 2: per batch — cross = logK · logQ^T (128x128, K=1024) via WMMA,
// fused energy -> score at store. grid = 32, block = 256.
// Operand chunks staged directly into f16 fragment layout.
// ============================================================================
__global__ void __launch_bounds__(256) k2_scores(
    const float* __restrict__ logK, const float* __restrict__ logQ,
    const float* __restrict__ k2, const float* __restrict__ q2,
    float* __restrict__ scores) {
  __shared__ v16h KA[8 * 32];     // logK chunk as A-frags [rt][lane]
  __shared__ v16h QB[8 * 32];     // logQ^T chunk as B-frags [ct][lane]
  _Float16* KAh = (_Float16*)KA;
  _Float16* QBh = (_Float16*)QB;

  const int tid = threadIdx.x, lane = tid & 31, wave = tid >> 5;
  const int b = blockIdx.x;
  const float* Kb = logK + (size_t)b * 128 * 1024;
  const float* Qb = logQ + (size_t)b * 128 * 1024;

  v8f acc[8] = {};
  for (int kc = 0; kc < 32; ++kc) {
    const int k0 = kc * 32;
    __syncthreads();
    for (int idx = tid; idx < 4096; idx += 256) {
      const int r = idx >> 5, c = idx & 31;    // r = matrix row (i or j), c = k
      int l, e;
      a_map(r & 15, c, l, e);                  // A[i][k] = logK[i][k]
      KAh[((r >> 4) * 32 + l) * 16 + e] = (_Float16)Kb[(size_t)r * 1024 + k0 + c];
      b_map(c, r & 15, l, e);                  // B[k][j] = logQ[j][k]
      QBh[((r >> 4) * 32 + l) * 16 + e] = (_Float16)Qb[(size_t)r * 1024 + k0 + c];
    }
    __syncthreads();
    const v16h a = KA[wave * 32 + lane];       // wave owns a row tile
#pragma unroll
    for (int ct = 0; ct < 8; ++ct)
      acc[ct] = WMMA_F32_F16(a, QB[ct * 32 + lane], acc[ct]);
  }

  // energy[b,i,j] = k2[i] + q2[j] - 2*cross; score = 1/(1+log1p(energy))
  const int h = (lane >> 4) & 1, n = lane & 15;
#pragma unroll
  for (int ct = 0; ct < 8; ++ct) {
    const int j = ct * 16 + n;
    const float qj = q2[b * 128 + j];
#pragma unroll
    for (int r = 0; r < 8; ++r) {
      const int i = wave * 16 + (h << 3) + r;
      const float e = fmaxf(k2[b * 128 + i] + qj - 2.f * acc[ct][r], 0.f);
      scores[((size_t)b * 128 + i) * 128 + j] = 1.f / (1.f + log1pf(e));
    }
  }
}

// ============================================================================
// Kernel 2b: softmax over K index i (axis -2), store transposed attn[b][j][i].
// grid = 32, block = 128 (thread = column j). Tiny: ~4 MB total traffic.
// ============================================================================
__global__ void __launch_bounds__(128) k2b_softmax(
    const float* __restrict__ scores, float* __restrict__ attn) {
  const int b = blockIdx.x, j = threadIdx.x;
  const float* S = scores + (size_t)b * 128 * 128;
  float mx = -1e30f;
  for (int i = 0; i < 128; ++i) mx = fmaxf(mx, S[i * 128 + j]);
  float sum = 0.f;
  for (int i = 0; i < 128; ++i) sum += __expf(S[i * 128 + j] - mx);
  const float inv = 1.f / sum;
  float* A = attn + ((size_t)b * 128 + j) * 128;
  for (int i = 0; i < 128; ++i) A[i] = __expf(S[i * 128 + j] - mx) * inv;
}

// ============================================================================
// Kernel 3a: mean_log[b][j][:] = attn[b][j][:] · logV[b] (128x1024, K=128).
// grid = 32 batches x 8 row tiles = 256, block = 256. attn row-block staged
// into f16 A-frag layout; logV B-frags read coalesced from L2.
// ============================================================================
__global__ void __launch_bounds__(256) k3a_meanlog(
    const float* __restrict__ attn, const float* __restrict__ logV,
    float* __restrict__ mean) {
  __shared__ v16h AA[4 * 32];                  // attn 16x128 as A-frags [kc][lane]
  _Float16* AAh = (_Float16*)AA;

  const int tid = threadIdx.x, lane = tid & 31, wave = tid >> 5;
  const int b = blockIdx.x >> 3;
  const int rt = blockIdx.x & 7;
  const float* Ab = attn + (size_t)b * 128 * 128;
  const float* Vb = logV + (size_t)b * 128 * 1024;

  for (int idx = tid; idx < 2048; idx += 256) {
    const int r = idx >> 7, c = idx & 127;     // r = local row (0..15), c = k
    int l, e; a_map(r, c & 31, l, e);
    AAh[((c >> 5) * 32 + l) * 16 + e] = (_Float16)Ab[(size_t)(rt * 16 + r) * 128 + c];
  }
  __syncthreads();

  v8f acc[8] = {};
#pragma unroll
  for (int kc = 0; kc < 4; ++kc) {
    if (kc < 3) __builtin_prefetch(Vb + (size_t)(kc + 1) * 32 * 1024, 0, 1);
    const v16h a = AA[kc * 32 + lane];
#pragma unroll
    for (int cc = 0; cc < 8; ++cc) {
      const v16h bb = frag_b_g(Vb, 1024, kc * 32, (wave * 8 + cc) * 16, lane);
      acc[cc] = WMMA_F32_F16(a, bb, acc[cc]);
    }
  }
  float* Mb = mean + (size_t)b * 128 * 1024;
#pragma unroll
  for (int cc = 0; cc < 8; ++cc)
    store_c(Mb, 1024, rt * 16, (wave * 8 + cc) * 16, lane, acc[cc]);
}

// ============================================================================
// Kernel 3b: out = exp_sym(mean_log) per matrix (Jacobi + WMMA reconstruct).
// grid = 4096, block = 256.
// ============================================================================
__global__ void __launch_bounds__(256) k3b_expsym(
    const float* __restrict__ mean, float* __restrict__ out) {
  __shared__ float SS[32 * 33], US[32 * 33];
  __shared__ v16h GA[2 * 32];
  __shared__ v16h UB[2 * 32];
  __shared__ float ev[32], fv[32];
  __shared__ float rc[16], rs[16];
  __shared__ int   pp[16], pq[16];
  _Float16* GAh = (_Float16*)GA;
  _Float16* UBh = (_Float16*)UB;

  const int tid = threadIdx.x, lane = tid & 31, wave = tid >> 5;
  const int mat = blockIdx.x;

  for (int idx = tid; idx < 1024; idx += 256)
    SS[(idx >> 5) * 33 + (idx & 31)] = mean[(size_t)mat * 1024 + idx];
  __syncthreads();

  jacobi32(SS, US, ev, pp, pq, rc, rs, tid);
  if (tid < 32) fv[tid] = __expf(ev[tid]);
  __syncthreads();

  for (int idx = tid; idx < 1024; idx += 256) {
    const int r = idx >> 5, c = idx & 31;
    const float u = US[r * 33 + c];
    int l, e;
    a_map(r & 15, c, l, e);
    GAh[((r >> 4) * 32 + l) * 16 + e] = (_Float16)(u * fv[c]);
    b_map(c, r & 15, l, e);
    UBh[((r >> 4) * 32 + l) * 16 + e] = (_Float16)u;
  }
  __syncthreads();

  if (wave < 4) {
    const int rt = wave >> 1, ct = wave & 1;
    const int n = lane & 15, h = (lane >> 4) & 1;
    v8f acc = {};
    acc = WMMA_F32_F16(GA[rt * 32 + lane], UB[ct * 32 + lane], acc);
#pragma unroll
    for (int r = 0; r < 8; ++r)
      out[(size_t)mat * 1024 + (rt * 16 + (h << 3) + r) * 32 + ct * 16 + n] = acc[r];
  }
}

// ============================================================================
extern "C" void kernel_launch(void* const* d_in, const int* in_sizes, int n_in,
                              void* d_out, int out_size, void* d_ws, size_t ws_size,
                              hipStream_t stream) {
  (void)in_sizes; (void)n_in; (void)out_size; (void)ws_size;
  const float* x  = (const float*)d_in[0];
  const float* Wq = (const float*)d_in[1];
  const float* Wk = (const float*)d_in[2];
  const float* Wv = (const float*)d_in[3];
  float* out = (float*)d_out;
  float* ws  = (float*)d_ws;

  const size_t NMAT = 4096, DD = 1024;             // 4096 matrices of 32x32
  float* logQ   = ws;                              // 4,194,304 f
  float* logK   = logQ + NMAT * DD;                // 4,194,304 f
  float* logV   = logK + NMAT * DD;                // 4,194,304 f
  float* mean   = logV + NMAT * DD;                // 4,194,304 f
  float* scores = mean + NMAT * DD;                //   524,288 f
  float* attn   = scores + (size_t)32 * 128 * 128; //   524,288 f
  float* q2     = attn + (size_t)32 * 128 * 128;   //      4096 f
  float* k2     = q2 + NMAT;                       //      4096 f

  k1_bimap_log<<<4096, 256, 0, stream>>>(x, Wq, Wk, Wv, logQ, logK, logV, q2, k2);
  k2_scores   <<<32,   256, 0, stream>>>(logK, logQ, k2, q2, scores);
  k2b_softmax <<<32,   128, 0, stream>>>(scores, attn);
  k3a_meanlog <<<256,  256, 0, stream>>>(attn, logV, mean);
  k3b_expsym  <<<4096, 256, 0, stream>>>(mean, out);
}
